// CausalSelfAttention_4853313044761
// MI455X (gfx1250) — matrix operand order repaired and checked by reading, not compiled, not run on hardware
//
#include <hip/hip_runtime.h>
#include <hip/hip_bf16.h>

typedef __attribute__((ext_vector_type(16))) __bf16 v16bf;
typedef __attribute__((ext_vector_type(8)))  __bf16 v8bf;
typedef __attribute__((ext_vector_type(4)))  __bf16 v4bf;
typedef __attribute__((ext_vector_type(8)))  float  v8f;

#define WMMA_BF16(A,B,C) \
  __builtin_amdgcn_wmma_f32_16x16x32_bf16(false,(A),false,(B),(short)0,(C),false,false)

// 16-bit A-fragment (16x32 MxK): lane m = lane&15;
// elems 0..15 = K[2*kA..2*kA+15] (contiguous), kA = (lane<16)?0:8.
static __device__ __forceinline__ v16bf make_afrag(const __bf16* row, int kA) {
  v8bf lo = *(const v8bf*)(row + 2 * kA);
  v8bf hi = *(const v8bf*)(row + 2 * kA + 8);
  v16bf a;
#pragma unroll
  for (int i = 0; i < 8; ++i) { a[i] = lo[i]; a[i + 8] = hi[i]; }
  return a;
}

static __device__ __forceinline__ float rowmax16(float t) {
  t = fmaxf(t, __shfl_xor(t, 1, 32));
  t = fmaxf(t, __shfl_xor(t, 2, 32));
  t = fmaxf(t, __shfl_xor(t, 4, 32));
  t = fmaxf(t, __shfl_xor(t, 8, 32));
  return t;
}
static __device__ __forceinline__ float rowsum16(float t) {
  t += __shfl_xor(t, 1, 32);
  t += __shfl_xor(t, 2, 32);
  t += __shfl_xor(t, 4, 32);
  t += __shfl_xor(t, 8, 32);
  return t;
}

// ---------------- fp32 -> bf16 conversion (x4 vectorized) ----------------
__global__ __launch_bounds__(256) void k_cvt4(const float4* __restrict__ in,
                                              __bf16* __restrict__ out, int n4) {
  int i = blockIdx.x * 256 + threadIdx.x;
  if (i < n4) {
    float4 v = in[i];
    v4bf o;
    o[0] = (__bf16)v.x; o[1] = (__bf16)v.y; o[2] = (__bf16)v.z; o[3] = (__bf16)v.w;
    *(v4bf*)(out + (size_t)i * 4) = o;
  }
}

// ---------------- fused QKV projection: q/k/v = x @ W^T + b ----------------
// One wave = 32(M) x 64(N) tile: 2 A-frags share 4 B-frags -> 8 WMMA / K-step.
// K-loop unrolled x2 with static ping-pong buffers: loads land directly in the
// buffer registers (no rotation movs); each WMMA group waits only on the loads
// issued one half-iteration earlier.
// Q,K stored [B,H,T,D]; V stored transposed [B,H,D,T] for the PV GEMM.
__global__ __launch_bounds__(256) void k_gemm_qkv(
    const __bf16* __restrict__ xb,
    const __bf16* __restrict__ Wq, const float* __restrict__ bq,
    const __bf16* __restrict__ Wk, const float* __restrict__ bk,
    const __bf16* __restrict__ Wv, const float* __restrict__ bv,
    __bf16* __restrict__ Qo, __bf16* __restrict__ Ko, __bf16* __restrict__ Vt)
{
  const int lane = threadIdx.x & 31;
  const int wave = threadIdx.x >> 5;
  const int task = blockIdx.x * 8 + wave;       // 0..6143
  const int mt  = task & 127;                   // 32-row tile of B*T=4096
  const int rem = task >> 7;                    // 0..47
  const int nc  = rem & 15;                     // 64-col chunk
  const int sel = rem >> 4;                     // 0:q 1:k 2:v

  const __bf16* W    = (sel == 0) ? Wq : (sel == 1) ? Wk : Wv;
  const float*  bias = (sel == 0) ? bq : (sel == 1) ? bk : bv;

  const int m0 = mt * 32, n0 = nc * 64;
  const int half = lane >> 4, l15 = lane & 15;
  const int kA = half ? 8 : 0;     // A-fragment K base
  const int kB = half ? 16 : 0;    // B-fragment K base

  v8f zero = {0.f,0.f,0.f,0.f,0.f,0.f,0.f,0.f};
  v8f acc0[4] = {zero, zero, zero, zero};
  v8f acc1[4] = {zero, zero, zero, zero};

  const __bf16* arow0 = xb + (size_t)(m0 + l15) * 1024;
  const __bf16* arow1 = arow0 + (size_t)16 * 1024;
  const __bf16* wbase = W + (size_t)(n0 + l15) * 1024 + kB;

  // buffer A: k = 0
  v16bf a0A = make_afrag(arow0, kA);
  v16bf a1A = make_afrag(arow1, kA);
  v16bf bA[4];
#pragma unroll
  for (int j = 0; j < 4; ++j) bA[j] = *(const v16bf*)(wbase + (size_t)j * 16 * 1024);

  for (int k = 0; k < 1024; k += 64) {
    // ---- load buffer B (k+32) while computing on buffer A (k) ----
    const int k1 = k + 32;                       // always < 1024
    v16bf a0B = make_afrag(arow0 + k1, kA);
    v16bf a1B = make_afrag(arow1 + k1, kA);
    v16bf bB[4];
#pragma unroll
    for (int j = 0; j < 4; ++j) bB[j] = *(const v16bf*)(wbase + (size_t)j * 16 * 1024 + k1);

#pragma unroll
    for (int j = 0; j < 4; ++j) acc0[j] = WMMA_BF16(a0A, bA[j], acc0[j]);
#pragma unroll
    for (int j = 0; j < 4; ++j) acc1[j] = WMMA_BF16(a1A, bA[j], acc1[j]);

    // ---- load buffer A (k+64) while computing on buffer B (k+32) ----
    const int k2 = (k + 64 < 1024) ? (k + 64) : 0;  // last iter: harmless reload
    a0A = make_afrag(arow0 + k2, kA);
    a1A = make_afrag(arow1 + k2, kA);
#pragma unroll
    for (int j = 0; j < 4; ++j) bA[j] = *(const v16bf*)(wbase + (size_t)j * 16 * 1024 + k2);

#pragma unroll
    for (int j = 0; j < 4; ++j) acc0[j] = WMMA_BF16(a0B, bB[j], acc0[j]);
#pragma unroll
    for (int j = 0; j < 4; ++j) acc1[j] = WMMA_BF16(a1B, bB[j], acc1[j]);
  }

#pragma unroll
  for (int s = 0; s < 2; ++s) {
    v8f* acc = s ? acc1 : acc0;
    const int ms = m0 + s * 16;
#pragma unroll
    for (int j = 0; j < 4; ++j) {
      const int c = n0 + j * 16 + l15;
      const float bb = bias[c];
      const int h = c >> 6, d = c & 63;
      const int t0 = ms + half * 8;             // rows t0..t0+7 in this lane
      const int b = t0 >> 11, tt0 = t0 & 2047;
      if (sel < 2) {
        __bf16* out = (sel == 0) ? Qo : Ko;
#pragma unroll
        for (int r = 0; r < 8; ++r)
          out[(((size_t)(b * 16 + h)) * 2048 + tt0 + r) * 64 + d] = (__bf16)(acc[j][r] + bb);
      } else {
        v8bf pack;
#pragma unroll
        for (int r = 0; r < 8; ++r) pack[r] = (__bf16)(acc[j][r] + bb);
        // Vt[(b,h,d)][t]: 8 contiguous t per lane -> one 16B store
        *(v8bf*)(Vt + (((size_t)(b * 16 + h)) * 64 + d) * 2048 + tt0) = pack;
      }
    }
  }
}

// ---------------- flash attention (causal, online softmax) ----------------
// One wave per 16-row Q strip; 32-key blocks; V B-frags loaded up front so the
// softmax VALU/shuffle work overlaps the V loads; P staged through wave-private
// LDS to convert WMMA C-layout -> A-layout.
__global__ __launch_bounds__(256) void k_flash(
    const __bf16* __restrict__ Qb, const __bf16* __restrict__ Kb,
    const __bf16* __restrict__ Vt, __bf16* __restrict__ yb)
{
  __shared__ __attribute__((aligned(64))) __bf16 lds[8][16][32];
  const int lane = threadIdx.x & 31;
  const int wave = threadIdx.x >> 5;
  const int wid  = blockIdx.x * 8 + wave;       // 0..4095
  const int qt = wid & 127;
  const int bh = wid >> 7;                      // 0..31 (b*16+h)
  const int q0 = qt * 16;
  const int half = lane >> 4, l15 = lane & 15;
  const int kA = half ? 8 : 0;
  const int kB = half ? 16 : 0;
  const float L2E = 1.44269504f;

  const __bf16* qrow = Qb + ((size_t)bh * 2048 + q0 + l15) * 64;
  v16bf aq0 = make_afrag(qrow, kA);             // d = 0..31
  v16bf aq1 = make_afrag(qrow + 32, kA);        // d = 32..63

  v8f zero = {0.f,0.f,0.f,0.f,0.f,0.f,0.f,0.f};
  v8f o[4] = {zero, zero, zero, zero};
  float rmax[8], rsum[8];
#pragma unroll
  for (int r = 0; r < 8; ++r) { rmax[r] = -1e30f; rsum[r] = 0.f; }

  const __bf16* kbp = Kb + (size_t)bh * 2048 * 64;
  const __bf16* vbp = Vt + (size_t)bh * 64 * 2048;

  for (int kb = 0; kb < q0 + 16; kb += 32) {
    // ---- issue K and V fragment loads together ----
    const __bf16* krow0 = kbp + (size_t)(kb + l15) * 64;
    const __bf16* krow1 = krow0 + 16 * 64;
    __builtin_prefetch(krow0 + 32 * 64, 0, 0);  // speculative next-block prefetch
    v16bf b0 = *(const v16bf*)(krow0 + kB);
    v16bf b1 = *(const v16bf*)(krow0 + 32 + kB);
    v16bf b2 = *(const v16bf*)(krow1 + kB);
    v16bf b3 = *(const v16bf*)(krow1 + 32 + kB);
    v16bf bv[4];
#pragma unroll
    for (int j = 0; j < 4; ++j) {
      // B[kk][d] = V[kk][d] = Vt[d][kk]: contiguous over kk
      bv[j] = *(const v16bf*)(vbp + (size_t)(j * 16 + l15) * 2048 + kb + kB);
    }

    // ---- S = Q @ K^T for 32 keys (two 16x16 tiles) ----
    v8f s0 = zero, s1 = zero;
    s0 = WMMA_BF16(aq0, b0, s0);
    s0 = WMMA_BF16(aq1, b1, s0);
    s1 = WMMA_BF16(aq0, b2, s1);
    s1 = WMMA_BF16(aq1, b3, s1);

    // ---- causal mask + scale + online softmax update (V loads in flight) ----
    const int kc0 = kb + l15, kc1 = kb + 16 + l15;
    float f[8];
#pragma unroll
    for (int r = 0; r < 8; ++r) {
      const int q = q0 + half * 8 + r;
      float v0 = s0[r] * 0.125f; if (kc0 > q) v0 = -1e30f;   // 1/sqrt(64)
      float v1 = s1[r] * 0.125f; if (kc1 > q) v1 = -1e30f;
      float mx = rowmax16(fmaxf(v0, v1));
      float nm = fmaxf(rmax[r], mx);
      float fr = exp2f((rmax[r] - nm) * L2E);
      float p0 = exp2f((v0 - nm) * L2E);
      float p1 = exp2f((v1 - nm) * L2E);
      rsum[r] = rsum[r] * fr + rowsum16(p0 + p1);
      rmax[r] = nm;
      f[r] = fr;
      lds[wave][half * 8 + r][l15]      = (__bf16)p0;  // P in [m][k] layout
      lds[wave][half * 8 + r][16 + l15] = (__bf16)p1;
    }
#pragma unroll
    for (int j = 0; j < 4; ++j)
#pragma unroll
      for (int r = 0; r < 8; ++r) o[j][r] *= f[r];

    // ---- O += P @ V ----
    v16bf ap = make_afrag(&lds[wave][l15][0], kA);     // A-frag of P from LDS
#pragma unroll
    for (int j = 0; j < 4; ++j) o[j] = WMMA_BF16(ap, bv[j], o[j]);
  }

  // ---- normalize and store y[b,t,c] ----
  const int b = bh >> 4, h = bh & 15;
#pragma unroll
  for (int r = 0; r < 8; ++r) rsum[r] = 1.f / rsum[r];
#pragma unroll
  for (int j = 0; j < 4; ++j) {
    const int c = h * 64 + j * 16 + l15;
#pragma unroll
    for (int r = 0; r < 8; ++r) {
      const int q = q0 + half * 8 + r;
      yb[((size_t)b * 2048 + q) * 1024 + c] = (__bf16)(o[j][r] * rsum[r]);
    }
  }
}

// ---------------- output projection: out = y @ Wp^T + bp (fp32 out) ----------------
// Same x2-unrolled ping-pong 32x64 wave tile as k_gemm_qkv.
__global__ __launch_bounds__(256) void k_gemm_proj(
    const __bf16* __restrict__ yb, const __bf16* __restrict__ Wp,
    const float* __restrict__ bp, float* __restrict__ out)
{
  const int lane = threadIdx.x & 31;
  const int wave = threadIdx.x >> 5;
  const int task = blockIdx.x * 8 + wave;       // 0..2047
  const int mt = task & 127;
  const int nc = task >> 7;
  const int m0 = mt * 32, n0 = nc * 64;
  const int half = lane >> 4, l15 = lane & 15;
  const int kA = half ? 8 : 0, kB = half ? 16 : 0;

  v8f zero = {0.f,0.f,0.f,0.f,0.f,0.f,0.f,0.f};
  v8f acc0[4] = {zero, zero, zero, zero};
  v8f acc1[4] = {zero, zero, zero, zero};

  const __bf16* arow0 = yb + (size_t)(m0 + l15) * 1024;
  const __bf16* arow1 = arow0 + (size_t)16 * 1024;
  const __bf16* wbase = Wp + (size_t)(n0 + l15) * 1024 + kB;

  v16bf a0A = make_afrag(arow0, kA);
  v16bf a1A = make_afrag(arow1, kA);
  v16bf bA[4];
#pragma unroll
  for (int j = 0; j < 4; ++j) bA[j] = *(const v16bf*)(wbase + (size_t)j * 16 * 1024);

  for (int k = 0; k < 1024; k += 64) {
    const int k1 = k + 32;
    v16bf a0B = make_afrag(arow0 + k1, kA);
    v16bf a1B = make_afrag(arow1 + k1, kA);
    v16bf bB[4];
#pragma unroll
    for (int j = 0; j < 4; ++j) bB[j] = *(const v16bf*)(wbase + (size_t)j * 16 * 1024 + k1);

#pragma unroll
    for (int j = 0; j < 4; ++j) acc0[j] = WMMA_BF16(a0A, bA[j], acc0[j]);
#pragma unroll
    for (int j = 0; j < 4; ++j) acc1[j] = WMMA_BF16(a1A, bA[j], acc1[j]);

    const int k2 = (k + 64 < 1024) ? (k + 64) : 0;
    a0A = make_afrag(arow0 + k2, kA);
    a1A = make_afrag(arow1 + k2, kA);
#pragma unroll
    for (int j = 0; j < 4; ++j) bA[j] = *(const v16bf*)(wbase + (size_t)j * 16 * 1024 + k2);

#pragma unroll
    for (int j = 0; j < 4; ++j) acc0[j] = WMMA_BF16(a0B, bB[j], acc0[j]);
#pragma unroll
    for (int j = 0; j < 4; ++j) acc1[j] = WMMA_BF16(a1B, bB[j], acc1[j]);
  }

#pragma unroll
  for (int s = 0; s < 2; ++s) {
    v8f* acc = s ? acc1 : acc0;
    const int ms = m0 + s * 16;
#pragma unroll
    for (int j = 0; j < 4; ++j) {
      const int c = n0 + j * 16 + l15;
      const float bb = bp[c];
#pragma unroll
      for (int r = 0; r < 8; ++r) {
        const int m = ms + half * 8 + r;
        out[(size_t)m * 1024 + c] = acc[j][r] + bb;
      }
    }
  }
}

extern "C" void kernel_launch(void* const* d_in, const int* in_sizes, int n_in,
                              void* d_out, int out_size, void* d_ws, size_t ws_size,
                              hipStream_t stream) {
  const float* x  = (const float*)d_in[0];
  const float* Wq = (const float*)d_in[1];
  const float* bq = (const float*)d_in[2];
  const float* Wk = (const float*)d_in[3];
  const float* bk = (const float*)d_in[4];
  const float* Wv = (const float*)d_in[5];
  const float* bv = (const float*)d_in[6];
  const float* Wp = (const float*)d_in[7];
  const float* bp = (const float*)d_in[8];

  char* ws = (char*)d_ws;
  size_t off = 0;
  const size_t SZ_X  = (size_t)4096 * 1024 * 2;        // bf16 x        : 8 MiB
  const size_t SZ_W  = (size_t)1024 * 1024 * 2;        // bf16 weight   : 2 MiB
  const size_t SZ_HD = (size_t)2 * 16 * 2048 * 64 * 2; // [B,H,T,D]     : 8 MiB

  __bf16* xb  = (__bf16*)(ws + off); off += SZ_X;
  __bf16* Wqb = (__bf16*)(ws + off); off += SZ_W;
  __bf16* Wkb = (__bf16*)(ws + off); off += SZ_W;
  __bf16* Wvb = (__bf16*)(ws + off); off += SZ_W;
  __bf16* Wpb = (__bf16*)(ws + off); off += SZ_W;
  __bf16* Qb  = (__bf16*)(ws + off); off += SZ_HD;
  __bf16* Kb  = (__bf16*)(ws + off); off += SZ_HD;
  __bf16* Vtb = (__bf16*)(ws + off); off += SZ_HD;
  __bf16* ybf = (__bf16*)(ws + off); off += SZ_X;      // total 48 MiB

  const int NX4 = 4096 * 1024 / 4;
  const int NW4 = 1024 * 1024 / 4;
  k_cvt4<<<(NX4 + 255) / 256, 256, 0, stream>>>((const float4*)x,  xb,  NX4);
  k_cvt4<<<(NW4 + 255) / 256, 256, 0, stream>>>((const float4*)Wq, Wqb, NW4);
  k_cvt4<<<(NW4 + 255) / 256, 256, 0, stream>>>((const float4*)Wk, Wkb, NW4);
  k_cvt4<<<(NW4 + 255) / 256, 256, 0, stream>>>((const float4*)Wv, Wvb, NW4);
  k_cvt4<<<(NW4 + 255) / 256, 256, 0, stream>>>((const float4*)Wp, Wpb, NW4);

  // 3 mats * 16 nchunks * 128 mtiles = 6144 wave-tasks / 8 waves = 768 blocks
  k_gemm_qkv<<<768, 256, 0, stream>>>(xb, Wqb, bq, Wkb, bk, Wvb, bv, Qb, Kb, Vtb);
  // 32 (b,h) * 128 q-tiles = 4096 wave-tasks / 8 = 512 blocks
  k_flash<<<512, 256, 0, stream>>>(Qb, Kb, Vtb, ybf);
  // 16 nchunks * 128 mtiles = 2048 wave-tasks / 8 = 256 blocks
  k_gemm_proj<<<256, 256, 0, stream>>>(ybf, Wpb, bp, (float*)d_out);
}
